// Fp8Linear_18081812316497
// MI455X (gfx1250) — 8-bit matrix operand order repaired and checked by reading, compiled, not run on hardware
//
#include <hip/hip_runtime.h>
#include <stdint.h>

typedef __attribute__((ext_vector_type(16))) int   v16i;
typedef __attribute__((ext_vector_type(8)))  float v8f;
typedef int v4i_g __attribute__((vector_size(4 * sizeof(int))));

#define FP8_E4M3_MAX 448.0f

#if __has_builtin(__builtin_amdgcn_global_load_async_to_lds_b128)
#define HAVE_ASYNC_LDS 1
#else
#define HAVE_ASYNC_LDS 0
#endif

__device__ __forceinline__ void wait_async0() {
#if __has_builtin(__builtin_amdgcn_s_wait_asynccnt)
  __builtin_amdgcn_s_wait_asynccnt(0);
#else
  asm volatile("s_wait_asynccnt 0" ::: "memory");
#endif
}

__device__ __forceinline__ unsigned pack4_fp8(float x0, float x1, float x2, float x3) {
#if __has_builtin(__builtin_amdgcn_cvt_pk_fp8_f32)
  int v = 0;
  v = __builtin_amdgcn_cvt_pk_fp8_f32(x0, x1, v, false);
  v = __builtin_amdgcn_cvt_pk_fp8_f32(x2, x3, v, true);
  return (unsigned)v;
#else
  auto one = [](float x) -> unsigned {
    union { float f; unsigned u; } c; c.f = x;
    unsigned s = (c.u >> 31) << 7;
    int e = (int)((c.u >> 23) & 0xffu) - 127 + 7;
    unsigned m = (c.u >> 20) & 7u;
    if (e <= 0) return s;
    if (e > 15) { e = 15; m = 6; }
    return s | ((unsigned)e << 3) | m;
  };
  return one(x0) | (one(x1) << 8) | (one(x2) << 16) | (one(x3) << 24);
#endif
}

__global__ __launch_bounds__(256)
void quant_act_kernel(const float* __restrict__ inp, const float* __restrict__ iscale_p,
                      unsigned* __restrict__ outq, long long n4) {
  const long long i = (long long)blockIdx.x * 256 + threadIdx.x;
  if (i >= n4) return;
  const float inv_is = 1.0f / iscale_p[0];
  const float4 x = *(const float4*)(inp + i * 4);
  const float q0 = fminf(fmaxf(x.x * inv_is, -FP8_E4M3_MAX), FP8_E4M3_MAX);
  const float q1 = fminf(fmaxf(x.y * inv_is, -FP8_E4M3_MAX), FP8_E4M3_MAX);
  const float q2 = fminf(fmaxf(x.z * inv_is, -FP8_E4M3_MAX), FP8_E4M3_MAX);
  const float q3 = fminf(fmaxf(x.w * inv_is, -FP8_E4M3_MAX), FP8_E4M3_MAX);
  outq[i] = pack4_fp8(q0, q1, q2, q3);
}

__global__ __launch_bounds__(256)
void quant_wt_kernel(const float* __restrict__ wq, unsigned* __restrict__ wt, int N, int K) {
  __shared__ unsigned char sT[64 * 64];
  const int k0 = blockIdx.x * 64;
  const int n0 = blockIdx.y * 64;
  const int tid = threadIdx.x;
#pragma unroll
  for (int j = 0; j < 4; ++j) {
    const int quad = tid + j * 256;
    const int kk = quad >> 4;
    const int nq = (quad & 15) * 4;
    const float4 x = *(const float4*)(wq + (long long)(k0 + kk) * N + (n0 + nq));
    *(unsigned*)&sT[kk * 64 + nq] = pack4_fp8(x.x, x.y, x.z, x.w);
  }
  __syncthreads();
#pragma unroll
  for (int j = 0; j < 4; ++j) {
    const int widx = tid + j * 256;
    const int n = widx >> 4;
    const int c = widx & 15;
    const unsigned d = (unsigned)sT[(c * 4 + 0) * 64 + n]
                     | ((unsigned)sT[(c * 4 + 1) * 64 + n] << 8)
                     | ((unsigned)sT[(c * 4 + 2) * 64 + n] << 16)
                     | ((unsigned)sT[(c * 4 + 3) * 64 + n] << 24);
    wt[(long long)(n0 + n) * (K >> 2) + (k0 >> 2) + c] = d;
  }
}

__global__ __launch_bounds__(256)
void fp8_gemm_pre_kernel(const unsigned* __restrict__ aq, const unsigned* __restrict__ wt,
                         const float* __restrict__ bias,
                         const float* __restrict__ wscale_p, const float* __restrict__ iscale_p,
                         float* __restrict__ out, int M, int N, int K) {
  __shared__ unsigned sA[2][128 * 32];
  __shared__ unsigned sB[2][128 * 32];

  const int tid   = threadIdx.x;
  const int lane  = tid & 31;
  const int wave  = tid >> 5;
  const int waveM = wave & 3;
  const int waveN = wave >> 2;
  const int half  = lane >> 4;
  const int l16   = lane & 15;
  const int K4    = K >> 2;

  const long long blockM = (long long)blockIdx.x * 128;
  const long long blockN = (long long)blockIdx.y * 128;

  v8f acc[2][4];
#pragma unroll
  for (int i = 0; i < 2; ++i)
#pragma unroll
    for (int j = 0; j < 4; ++j) {
      v8f z = {0.f, 0.f, 0.f, 0.f, 0.f, 0.f, 0.f, 0.f};
      acc[i][j] = z;
    }

  auto stage = [&](int k0d, int buf) {
#pragma unroll
    for (int j = 0; j < 4; ++j) {
      const int quad = tid + j * 256;
      const int row  = quad >> 3;
      const int q4   = (quad & 7) * 4;
      const unsigned* ga = aq + (blockM + row) * K4 + k0d + q4;
      const unsigned* gb = wt + (blockN + row) * K4 + k0d + q4;
#if HAVE_ASYNC_LDS
      __builtin_amdgcn_global_load_async_to_lds_b128(
          (v4i_g*)ga, (v4i_g*)&sA[buf][row * 32 + q4], 0, 0);
      __builtin_amdgcn_global_load_async_to_lds_b128(
          (v4i_g*)gb, (v4i_g*)&sB[buf][row * 32 + q4], 0, 0);
#else
      *(uint4*)&sA[buf][row * 32 + q4] = *(const uint4*)ga;
      *(uint4*)&sB[buf][row * 32 + q4] = *(const uint4*)gb;
#endif
    }
  };

  stage(0, 0);
#if HAVE_ASYNC_LDS
  wait_async0();
#endif
  __syncthreads();

  const int nSteps = K4 / 32;
  int cur = 0;
  for (int s = 0; s < nSteps; ++s) {
    if (s + 1 < nSteps) stage((s + 1) * 32, cur ^ 1);

    v16i afrag[2];
#pragma unroll
    for (int mi = 0; mi < 2; ++mi) {
      const unsigned* base = &sA[cur][(waveM * 32 + mi * 16 + l16) * 32];
#pragma unroll
      for (int h = 0; h < 2; ++h)
#pragma unroll
        for (int p = 0; p < 4; ++p) {
          const uint2 d = *(const uint2*)(base + h * 16 + (p >> 1) * 8 + half * 4 + (p & 1) * 2);
          afrag[mi][h * 8 + p * 2 + 0] = (int)d.x;
          afrag[mi][h * 8 + p * 2 + 1] = (int)d.y;
        }
    }
    v16i bfrag[4];
#pragma unroll
    for (int ni = 0; ni < 4; ++ni) {
      const unsigned* base = &sB[cur][(waveN * 64 + ni * 16 + l16) * 32];
#pragma unroll
      for (int g = 0; g < 4; ++g) {
        const uint4 d = *(const uint4*)(base + g * 8 + half * 4);
        bfrag[ni][g * 4 + 0] = (int)d.x;
        bfrag[ni][g * 4 + 1] = (int)d.y;
        bfrag[ni][g * 4 + 2] = (int)d.z;
        bfrag[ni][g * 4 + 3] = (int)d.w;
      }
    }

#pragma unroll
    for (int mi = 0; mi < 2; ++mi)
#pragma unroll
      for (int ni = 0; ni < 4; ++ni)
        acc[mi][ni] = __builtin_amdgcn_wmma_f32_16x16x128_fp8_fp8(
            afrag[mi], bfrag[ni], (short)0, acc[mi][ni], false, false);

#if HAVE_ASYNC_LDS
    wait_async0();
#endif
    __syncthreads();
    cur ^= 1;
  }

  const float oscale = iscale_p[0] * wscale_p[0];
#pragma unroll
  for (int ni = 0; ni < 4; ++ni) {
    const long long col = blockN + waveN * 64 + ni * 16 + l16;
    const float bv = bias[col];
#pragma unroll
    for (int mi = 0; mi < 2; ++mi) {
      const long long rowBase = blockM + waveM * 32 + mi * 16 + half * 8;
#pragma unroll
      for (int r = 0; r < 8; ++r)
        out[(rowBase + r) * N + col] = acc[mi][ni][r] * oscale + bv;
    }
  }
}

__global__ __launch_bounds__(256)
void fp8_linear_fused_kernel(const float* __restrict__ inp, const float* __restrict__ wq,
                             const float* __restrict__ bias,
                             const float* __restrict__ wscale_p, const float* __restrict__ iscale_p,
                             float* __restrict__ out, int M, int N, int K) {
  __shared__ unsigned sA[128 * 32];
  __shared__ unsigned sB[128 * 32];

  const int tid   = threadIdx.x;
  const int lane  = tid & 31;
  const int wave  = tid >> 5;
  const int waveM = wave & 3;
  const int waveN = wave >> 2;
  const int half  = lane >> 4;
  const int l16   = lane & 15;

  const long long blockM = (long long)blockIdx.x * 128;
  const long long blockN = (long long)blockIdx.y * 128;
  const float inv_is = 1.0f / iscale_p[0];

  v8f acc[2][4];
#pragma unroll
  for (int i = 0; i < 2; ++i)
#pragma unroll
    for (int j = 0; j < 4; ++j) {
      v8f z = {0.f, 0.f, 0.f, 0.f, 0.f, 0.f, 0.f, 0.f};
      acc[i][j] = z;
    }

  for (int k0 = 0; k0 < K; k0 += 128) {
#pragma unroll
    for (int j = 0; j < 16; ++j) {
      const int idx = tid + j * 256;
      const int row = idx >> 5;
      const int c   = idx & 31;
      const float4 x = *(const float4*)(inp + (blockM + row) * K + (k0 + c * 4));
      const float q0 = fminf(fmaxf(x.x * inv_is, -FP8_E4M3_MAX), FP8_E4M3_MAX);
      const float q1 = fminf(fmaxf(x.y * inv_is, -FP8_E4M3_MAX), FP8_E4M3_MAX);
      const float q2 = fminf(fmaxf(x.z * inv_is, -FP8_E4M3_MAX), FP8_E4M3_MAX);
      const float q3 = fminf(fmaxf(x.w * inv_is, -FP8_E4M3_MAX), FP8_E4M3_MAX);
      sA[row * 32 + c] = pack4_fp8(q0, q1, q2, q3);
    }
#pragma unroll
    for (int j = 0; j < 16; ++j) {
      const int idx = tid + j * 256;
      const int n   = idx & 127;
      const int c   = idx >> 7;
      const float* wp = wq + (long long)(k0 + c * 4) * N + (blockN + n);
      sB[n * 32 + c] = pack4_fp8(wp[0], wp[(long long)N], wp[2ll * N], wp[3ll * N]);
    }
    __syncthreads();

    v16i afrag[2];
#pragma unroll
    for (int mi = 0; mi < 2; ++mi) {
      const unsigned* base = &sA[(waveM * 32 + mi * 16 + l16) * 32];
#pragma unroll
      for (int h = 0; h < 2; ++h)
#pragma unroll
        for (int p = 0; p < 4; ++p) {
          const uint2 d = *(const uint2*)(base + h * 16 + (p >> 1) * 8 + half * 4 + (p & 1) * 2);
          afrag[mi][h * 8 + p * 2 + 0] = (int)d.x;
          afrag[mi][h * 8 + p * 2 + 1] = (int)d.y;
        }
    }
    v16i bfrag[4];
#pragma unroll
    for (int ni = 0; ni < 4; ++ni) {
      const unsigned* base = &sB[(waveN * 64 + ni * 16 + l16) * 32];
#pragma unroll
      for (int g = 0; g < 4; ++g) {
        const uint4 d = *(const uint4*)(base + g * 8 + half * 4);
        bfrag[ni][g * 4 + 0] = (int)d.x;
        bfrag[ni][g * 4 + 1] = (int)d.y;
        bfrag[ni][g * 4 + 2] = (int)d.z;
        bfrag[ni][g * 4 + 3] = (int)d.w;
      }
    }
#pragma unroll
    for (int mi = 0; mi < 2; ++mi)
#pragma unroll
      for (int ni = 0; ni < 4; ++ni)
        acc[mi][ni] = __builtin_amdgcn_wmma_f32_16x16x128_fp8_fp8(
            afrag[mi], bfrag[ni], (short)0, acc[mi][ni], false, false);
    __syncthreads();
  }

  const float oscale = iscale_p[0] * wscale_p[0];
#pragma unroll
  for (int ni = 0; ni < 4; ++ni) {
    const long long col = blockN + waveN * 64 + ni * 16 + l16;
    const float bv = bias[col];
#pragma unroll
    for (int mi = 0; mi < 2; ++mi) {
      const long long rowBase = blockM + waveM * 32 + mi * 16 + half * 8;
#pragma unroll
      for (int r = 0; r < 8; ++r)
        out[(rowBase + r) * N + col] = acc[mi][ni][r] * oscale + bv;
    }
  }
}

extern "C" void kernel_launch(void* const* d_in, const int* in_sizes, int n_in,
                              void* d_out, int out_size, void* d_ws, size_t ws_size,
                              hipStream_t stream) {
  const float* inp    = (const float*)d_in[0];
  const float* wq     = (const float*)d_in[1];
  const float* bias   = (const float*)d_in[2];
  const float* wscale = (const float*)d_in[3];
  const float* iscale = (const float*)d_in[4];
  float* out = (float*)d_out;

  const int N = in_sizes[2];
  const int K = (int)((long long)in_sizes[1] / N);
  const int M = (int)((long long)in_sizes[0] / K);

  const size_t needA = (size_t)M * (size_t)K;
  const size_t needW = (size_t)N * (size_t)K;

  dim3 gemmGrid((M + 127) / 128, (N + 127) / 128);

  if (ws_size >= needA + needW) {
    unsigned* aq = (unsigned*)d_ws;
    unsigned* wt = (unsigned*)((char*)d_ws + needA);
    const long long n4 = (long long)M * K / 4;
    quant_act_kernel<<<dim3((unsigned)((n4 + 255) / 256)), dim3(256), 0, stream>>>(
        inp, iscale, aq, n4);
    quant_wt_kernel<<<dim3((K + 63) / 64, (N + 63) / 64), dim3(256), 0, stream>>>(
        wq, wt, N, K);
    fp8_gemm_pre_kernel<<<gemmGrid, dim3(256), 0, stream>>>(
        aq, wt, bias, wscale, iscale, out, M, N, K);
  } else {
    fp8_linear_fused_kernel<<<gemmGrid, dim3(256), 0, stream>>>(
        inp, wq, bias, wscale, iscale, out, M, N, K);
  }
}
